// FFTFrequencyBranch_11613591568827
// MI455X (gfx1250) — matrix operand order repaired and checked by reading, not compiled, not run on hardware
//
#include <hip/hip_runtime.h>
#include <hip/hip_bf16.h>
#include <math.h>

// ---------------------------------------------------------------------------
// Types for CDNA5 WMMA (wave32, 16x16x32 f16 -> f32) and TDM descriptors
// ---------------------------------------------------------------------------
typedef _Float16 v8h  __attribute__((ext_vector_type(8)));
typedef _Float16 v16h __attribute__((ext_vector_type(16)));
typedef float    v8f  __attribute__((ext_vector_type(8)));
typedef unsigned int u32x4 __attribute__((ext_vector_type(4)));
typedef int          i32x8 __attribute__((ext_vector_type(8)));
typedef int          i32x4 __attribute__((ext_vector_type(4)));

#define BN_EPS   1e-5f
#define NORM_EPS 1e-8f

__device__ __forceinline__ v8h zero8h() {
    v8h r;
#pragma unroll
    for (int i = 0; i < 8; ++i) r[i] = (_Float16)0.0f;
    return r;
}
__device__ __forceinline__ v8f zero8f() {
    v8f r;
#pragma unroll
    for (int i = 0; i < 8; ++i) r[i] = 0.0f;
    return r;
}
__device__ __forceinline__ v8h load8h(const _Float16* p) {
    return *(const v8h*)p;
}
__device__ __forceinline__ v16h mk16(v8h lo, v8h hi) {
    v16h r;
#pragma unroll
    for (int i = 0; i < 8; ++i) { r[i] = lo[i]; r[i + 8] = hi[i]; }
    return r;
}
__device__ __forceinline__ v8f wmma_f16(v16h a, v16h b, v8f c) {
    return __builtin_amdgcn_wmma_f32_16x16x32_f16(false, a, false, b,
                                                  (short)0, c, false, false);
}

// ---------------------------------------------------------------------------
// Tensor Data Mover: 2D tile DMA global -> LDS (D# per CDNA5 ISA §8.3/8.4).
// tile_d0 / stride_d0 are in 4-byte units (data_size = 4B). Tensor dims are
// oversized so no OOB clipping occurs (tile start is the descriptor base).
// This toolchain exposes the 6-arg builtin (g0, g1, g2, g3, g4, cpol).
// ---------------------------------------------------------------------------
__device__ __forceinline__ void tdm_load_2d(const void* gaddr, unsigned lds_off,
                                            unsigned tile_d0, unsigned tile_d1,
                                            unsigned stride_d0) {
    unsigned long long ga = (unsigned long long)(uintptr_t)gaddr;
    u32x4 g0;
    g0[0] = 1u;                                        // count=1, user D#
    g0[1] = lds_off;                                   // lds_addr (bytes)
    g0[2] = (unsigned)(ga & 0xffffffffu);              // global_addr[31:0]
    g0[3] = (unsigned)((ga >> 32) & 0x01ffffffu)       // global_addr[56:32]
          | (2u << 30);                                // type = 2 ("image")
    const unsigned td0 = 0x00100000u;                  // huge tensor_dim0
    const unsigned td1 = 0x00100000u;                  // huge tensor_dim1
    i32x8 g1;
    g1[0] = (int)(2u << 16);                           // data_size=2 -> 4B elems
    g1[1] = (int)((td0 & 0xffffu) << 16);              // tensor_dim0[15:0]
    g1[2] = (int)((td0 >> 16) | ((td1 & 0xffffu) << 16));
    g1[3] = (int)((td1 >> 16) | (tile_d0 << 16));      // tile_dim0
    g1[4] = (int)(tile_d1 & 0xffffu);                  // tile_dim1 (tile_dim2=0)
    g1[5] = (int)stride_d0;                            // tensor_dim0_stride lo32
    g1[6] = 0;                                         // stride hi / dim1_stride
    g1[7] = 0;
    i32x4 z4 = {0, 0, 0, 0};                           // 2D: groups 2/3 unused
    i32x8 z8 = {0, 0, 0, 0, 0, 0, 0, 0};
    __builtin_amdgcn_tensor_load_to_lds(g0, g1, z4, z4, z8, 0);
}

// ---------------------------------------------------------------------------
// Kernel 0: weight repack (f32 -> f16, WMMA-B friendly [tap][oc][cin]) + zero
// ---------------------------------------------------------------------------
__global__ void prep_kernel(const float* __restrict__ w2, const float* __restrict__ w3,
                            const float* __restrict__ f1, const float* __restrict__ f2,
                            _Float16* __restrict__ w2p, _Float16* __restrict__ w3p,
                            _Float16* __restrict__ f1h, _Float16* __restrict__ f2h,
                            float* __restrict__ stats, float* __restrict__ feat) {
    int i = blockIdx.x * blockDim.x + threadIdx.x;
    if (i < 9 * 64 * 32) {           // conv2: (64,32,3,3) -> [tap][oc][cin]
        int tap = i / (64 * 32), r = i % (64 * 32);
        int oc = r / 32, cin = r % 32;
        int ky = tap / 3, kx = tap % 3;
        w2p[i] = (_Float16)w2[((oc * 32 + cin) * 3 + ky) * 3 + kx];
    }
    if (i < 9 * 128 * 64) {          // conv3: (128,64,3,3) -> [tap][oc][cin]
        int tap = i / (128 * 64), r = i % (128 * 64);
        int oc = r / 64, cin = r % 64;
        int ky = tap / 3, kx = tap % 3;
        w3p[i] = (_Float16)w3[((oc * 64 + cin) * 3 + ky) * 3 + kx];
    }
    if (i < 256 * 128) f1h[i] = (_Float16)f1[i];   // fc1_w (256,128) row-major
    if (i < 128 * 256) f2h[i] = (_Float16)f2[i];   // fc2_w (128,256) row-major
    if (i < 128)  stats[i] = 0.0f;                 // per-image sum / sumsq
    if (i < 8192) feat[i]  = 0.0f;                 // (64,128) GAP accumulator
}

// ---------------------------------------------------------------------------
// Kernel 1: grayscale + row FFT (512-pt radix-2 DIT in LDS), one block per row
// ---------------------------------------------------------------------------
__global__ void fft_rows_kernel(const float* __restrict__ x, float2* __restrict__ F) {
    __shared__ float2 s[512];
    int row = blockIdx.x;                 // 0 .. 64*512-1
    int b = row >> 9, r = row & 511;
    const float* xb = x + (size_t)b * 3 * 512 * 512 + (size_t)r * 512;
    int t = threadIdx.x;                  // 256 threads
    for (int i = t; i < 512; i += 256) {
        float g = 0.299f * xb[i] + 0.587f * xb[i + 512 * 512]
                + 0.114f * xb[i + 2 * 512 * 512];
        int j = (int)(__brev((unsigned)i) >> 23);   // 9-bit bit-reverse
        s[j] = make_float2(g, 0.0f);
    }
    __syncthreads();
#pragma unroll
    for (int len = 2; len <= 512; len <<= 1) {
        int half = len >> 1;
        int blk = t / half, pos = t % half;
        int i0 = blk * len + pos, i1 = i0 + half;
        float ang = -6.2831853071795864f * (float)pos / (float)len;
        float sw, cw; __sincosf(ang, &sw, &cw);
        float2 u = s[i0], v = s[i1];
        float2 wv = make_float2(v.x * cw - v.y * sw, v.x * sw + v.y * cw);
        s[i0] = make_float2(u.x + wv.x, u.y + wv.y);
        s[i1] = make_float2(u.x - wv.x, u.y - wv.y);
        __syncthreads();
    }
    float2* o = F + ((size_t)b * 512 + r) * 512;
    for (int i = t; i < 512; i += 256) o[i] = s[i];
}

// ---------------------------------------------------------------------------
// Kernel 2: column FFT + fftshift + log1p|F| + per-image sum/sumsq (atomics)
// ---------------------------------------------------------------------------
__global__ void fft_cols_kernel(const float2* __restrict__ F,
                                float* __restrict__ logm, float* __restrict__ stats) {
    __shared__ float2 s[512];
    __shared__ float rs[256], rq[256];
    int col = blockIdx.x;
    int b = col >> 9, c = col & 511;
    const float2* Fb = F + (size_t)b * 512 * 512 + c;
    int t = threadIdx.x;
    for (int i = t; i < 512; i += 256) {
        int j = (int)(__brev((unsigned)i) >> 23);
        s[j] = Fb[(size_t)i * 512];
    }
    __syncthreads();
#pragma unroll
    for (int len = 2; len <= 512; len <<= 1) {
        int half = len >> 1;
        int blk = t / half, pos = t % half;
        int i0 = blk * len + pos, i1 = i0 + half;
        float ang = -6.2831853071795864f * (float)pos / (float)len;
        float sw, cw; __sincosf(ang, &sw, &cw);
        float2 u = s[i0], v = s[i1];
        float2 wv = make_float2(v.x * cw - v.y * sw, v.x * sw + v.y * cw);
        s[i0] = make_float2(u.x + wv.x, u.y + wv.y);
        s[i1] = make_float2(u.x - wv.x, u.y - wv.y);
        __syncthreads();
    }
    float sum = 0.0f, sq = 0.0f;
    int c2 = (c + 256) & 511;
    float* lb = logm + (size_t)b * 512 * 512;
    for (int i = t; i < 512; i += 256) {
        float2 v = s[i];
        float lm = log1pf(sqrtf(v.x * v.x + v.y * v.y));
        int r2 = (i + 256) & 511;
        lb[(size_t)r2 * 512 + c2] = lm;   // fftshift folded into the store
        sum += lm; sq += lm * lm;
    }
    rs[t] = sum; rq[t] = sq;
    __syncthreads();
    for (int off = 128; off > 0; off >>= 1) {
        if (t < off) { rs[t] += rs[t + off]; rq[t] += rq[t + off]; }
        __syncthreads();
    }
    if (t == 0) {
        atomicAdd(&stats[2 * b + 0], rs[0]);
        atomicAdd(&stats[2 * b + 1], rq[0]);
    }
}

// ---------------------------------------------------------------------------
// Kernel 3: normalize + conv1(1->32,5x5,s2,p2) + BN + ReLU + 2x2 maxpool
//           output NHWC f16 (B,128,128,32). LDS-tiled direct conv (cin==1).
// ---------------------------------------------------------------------------
__global__ void conv1_kernel(const float* __restrict__ logm, const float* __restrict__ stats,
                             const float* __restrict__ w, const float* __restrict__ bias,
                             const float* __restrict__ g, const float* __restrict__ be,
                             const float* __restrict__ mn, const float* __restrict__ vr,
                             _Float16* __restrict__ h1) {
    __shared__ float tile[67 * 67];
    __shared__ float wsm[32 * 25];
    __shared__ float sc[32], offs[32];
    int blk = blockIdx.x;                  // 64 img * 64 tiles
    int b = blk >> 6, t6 = blk & 63;
    int TY = (t6 >> 3) * 16, TX = (t6 & 7) * 16;   // pooled-tile origin
    int tid = threadIdx.x;

    const float invN = 1.0f / (512.0f * 512.0f);
    float mu  = stats[2 * b + 0] * invN;
    float var = stats[2 * b + 1] * invN - mu * mu;
    float isd = 1.0f / (sqrtf(fmaxf(var, 0.0f)) + NORM_EPS);

    const float* lb = logm + (size_t)b * 512 * 512;
    int oy0 = 4 * TY - 2, ox0 = 4 * TX - 2;        // input-window origin
    for (int idx = tid; idx < 67 * 67; idx += 256) {
        int iy = oy0 + idx / 67, ix = ox0 + idx % 67;
        float v = 0.0f;
        if ((unsigned)iy < 512u && (unsigned)ix < 512u)
            v = (lb[(size_t)iy * 512 + ix] - mu) * isd;
        tile[idx] = v;
    }
    for (int idx = tid; idx < 32 * 25; idx += 256) wsm[idx] = w[idx];
    if (tid < 32) {
        float s = g[tid] * rsqrtf(vr[tid] + BN_EPS);
        sc[tid]   = s;
        offs[tid] = be[tid] - mn[tid] * s + bias[tid] * s;
    }
    __syncthreads();

    int py = tid >> 4, px = tid & 15;              // local pooled coords
    _Float16* out = h1 + (((size_t)(b * 128 + TY + py) * 128) + (TX + px)) * 32;
#pragma unroll 4
    for (int oc = 0; oc < 32; ++oc) {
        float best = 0.0f;                         // ReLU outputs are >= 0
#pragma unroll
        for (int d = 0; d < 4; ++d) {
            int oyl = 2 * py + (d >> 1), oxl = 2 * px + (d & 1);
            float acc = 0.0f;
#pragma unroll
            for (int ky = 0; ky < 5; ++ky)
#pragma unroll
                for (int kx = 0; kx < 5; ++kx)
                    acc += tile[(2 * oyl + ky) * 67 + (2 * oxl + kx)]
                         * wsm[oc * 25 + ky * 5 + kx];
            best = fmaxf(best, fmaxf(acc * sc[oc] + offs[oc], 0.0f));
        }
        out[oc] = (_Float16)best;
    }
}

// ---------------------------------------------------------------------------
// Kernel 4: conv2(32->64,3x3,s2,p1) implicit GEMM via WMMA + BN + ReLU + pool
//           h1 NHWC f16 (B,128,128,32) -> h2 NHWC f16 (B,32,32,64)
//           Block = 8 waves sharing one oc-tile; weights staged by the TDM.
// ---------------------------------------------------------------------------
__global__ void conv2_kernel(const _Float16* __restrict__ h1, const _Float16* __restrict__ w2p,
                             const float* __restrict__ bias, const float* __restrict__ g,
                             const float* __restrict__ be, const float* __restrict__ mn,
                             const float* __restrict__ vr, _Float16* __restrict__ h2) {
    __shared__ __align__(16) _Float16 w2s[9 * 16 * 32];   // 9216 B weight slice
    int tid = threadIdx.x, lane = tid & 31, wv = tid >> 5;
    int blk = blockIdx.x;                          // 8192 blocks
    int b = blk >> 7, rem = blk & 127;
    int nt = rem >> 5, sg = rem & 31;              // oc-tile, spatial group
    int st = sg * 8 + wv;                          // spatial tile 0..255
    int Y = (st >> 4) << 2, X = (st & 15) << 2;    // conv-space 4x4 block origin
    int m = lane & 15, hiHalf = lane >> 4;
    int oy = Y + (m >> 2), ox = X + (m & 3);       // this lane's A row (pixel)
    int n = nt * 16 + m;                           // this lane's D column (oc)
    int c0 = hiHalf ? 8 : 0;                       // A K-offsets per ISA layout
    int k0 = hiHalf ? 8 : 0;                       // B K-offsets per ISA layout

    // TDM: DMA this block's weight slice [9 taps][16 oc][32 cin] f16 into LDS.
    // Global tile: 9 rows of 16*32*2 = 1024 B (256 dwords), row stride
    // 64*32*2 = 4096 B (1024 dwords), base = w2p + nt*16*32 halves.
    if (wv == 0) {
        unsigned lds_off = (unsigned)(uintptr_t)(void*)w2s;
        tdm_load_2d(w2p + (size_t)nt * 16 * 32, lds_off,
                    /*tile_d0=*/256, /*tile_d1=*/9, /*stride_d0=*/1024);
        __builtin_amdgcn_s_wait_tensorcnt(0);
    }
    __syncthreads();

    v8f acc = zero8f();
    v8h zh = zero8h();
#pragma unroll
    for (int ky = 0; ky < 3; ++ky) {
        int iy = 2 * oy + ky - 1;
        bool yok = (unsigned)iy < 128u;
#pragma unroll
        for (int kx = 0; kx < 3; ++kx) {
            int ix = 2 * ox + kx - 1;
            bool ok = yok && (unsigned)ix < 128u;
            size_t abase = ok ? ((((size_t)(b * 128 + iy)) * 128 + ix) * 32 + c0) : 0;
            if (ky < 2)   // prefetch next input row (global_prefetch_b8 path)
                __builtin_prefetch((const void*)(h1 + abase + 128 * 32), 0, 1);
            v8h alo = load8h(h1 + abase);
            v8h ahi = load8h(h1 + abase + 16);
            alo = ok ? alo : zh;
            ahi = ok ? ahi : zh;
            int tap = ky * 3 + kx;
            const _Float16* bp = &w2s[(tap * 16 + m) * 32 + k0];   // LDS B-frag
            v8h blo = load8h(bp);
            v8h bhi = load8h(bp + 16);
            acc = wmma_f16(mk16(alo, ahi), mk16(blo, bhi), acc);
        }
    }
    // Epilogue: BN + ReLU + 2x2 maxpool inside the D fragment.
    float s = g[n] * rsqrtf(vr[n] + BN_EPS);
    float o = be[n] - mn[n] * s + bias[n] * s;
    float y[8];
#pragma unroll
    for (int i = 0; i < 8; ++i) y[i] = fmaxf(acc[i] * s + o, 0.0f);
    float p0 = fmaxf(fmaxf(y[0], y[1]), fmaxf(y[4], y[5]));   // pooled px=0
    float p1 = fmaxf(fmaxf(y[2], y[3]), fmaxf(y[6], y[7]));   // pooled px=1
    int pY = (Y >> 1) + hiHalf, pX = X >> 1;
    _Float16* op = h2 + (((size_t)(b * 32 + pY)) * 32 + pX) * 64 + n;
    op[0]  = (_Float16)p0;
    op[64] = (_Float16)p1;
}

// ---------------------------------------------------------------------------
// Kernel 5: conv3(64->128,3x3,s2,p1) WMMA + BN + ReLU + global-average-pool
//           h2 (B,32,32,64) -> feat (B,128) f32. TDM-staged weights in LDS.
// ---------------------------------------------------------------------------
__global__ void conv3_kernel(const _Float16* __restrict__ h2, const _Float16* __restrict__ w3p,
                             const float* __restrict__ bias, const float* __restrict__ g,
                             const float* __restrict__ be, const float* __restrict__ mn,
                             const float* __restrict__ vr, float* __restrict__ feat) {
    __shared__ __align__(16) _Float16 w3s[9 * 16 * 64];   // 18432 B weight slice
    int tid = threadIdx.x, lane = tid & 31, wv = tid >> 5;
    int blk = blockIdx.x;                          // 1024 blocks
    int b = blk >> 4, rem = blk & 15;
    int nt = rem >> 1, sg = rem & 1;
    int st = sg * 8 + wv;                          // spatial tile 0..15
    int Y = (st >> 2) << 2, X = (st & 3) << 2;
    int m = lane & 15, hiHalf = lane >> 4;
    int oy = Y + (m >> 2), ox = X + (m & 3);
    int n = nt * 16 + m;
    int k0 = hiHalf ? 8 : 0;

    // TDM: weight slice [9 taps][16 oc][64 cin] f16: 9 rows of 2048 B
    // (512 dwords), row stride 128*64*2 = 16384 B (4096 dwords).
    if (wv == 0) {
        unsigned lds_off = (unsigned)(uintptr_t)(void*)w3s;
        tdm_load_2d(w3p + (size_t)nt * 16 * 64, lds_off,
                    /*tile_d0=*/512, /*tile_d1=*/9, /*stride_d0=*/4096);
        __builtin_amdgcn_s_wait_tensorcnt(0);
    }
    __syncthreads();

    v8f acc = zero8f();
    v8h zh = zero8h();
#pragma unroll
    for (int ky = 0; ky < 3; ++ky) {
        int iy = 2 * oy + ky - 1;
        bool yok = (unsigned)iy < 32u;
#pragma unroll
        for (int kx = 0; kx < 3; ++kx) {
            int ix = 2 * ox + kx - 1;
            bool ok = yok && (unsigned)ix < 32u;
            int tap = ky * 3 + kx;
#pragma unroll
            for (int cb = 0; cb < 64; cb += 32) {
                size_t abase = ok ? ((((size_t)(b * 32 + iy)) * 32 + ix) * 64
                                     + cb + (hiHalf ? 8 : 0)) : 0;
                v8h alo = load8h(h2 + abase);
                v8h ahi = load8h(h2 + abase + 16);
                alo = ok ? alo : zh;
                ahi = ok ? ahi : zh;
                const _Float16* bp = &w3s[(tap * 16 + m) * 64 + cb + k0]; // LDS
                v8h blo = load8h(bp);
                v8h bhi = load8h(bp + 16);
                acc = wmma_f16(mk16(alo, ahi), mk16(blo, bhi), acc);
            }
        }
    }
    float s = g[n] * rsqrtf(vr[n] + BN_EPS);
    float o = be[n] - mn[n] * s + bias[n] * s;
    float sum = 0.0f;
#pragma unroll
    for (int i = 0; i < 8; ++i) sum += fmaxf(acc[i] * s + o, 0.0f);
    atomicAdd(&feat[b * 128 + n], sum * (1.0f / 256.0f));   // mean over 16x16
}

// ---------------------------------------------------------------------------
// Kernel 6: FC1(128->256)+ReLU, FC2(256->128)+ReLU. One block, 8 waves, WMMA.
// ---------------------------------------------------------------------------
__global__ void fc_kernel(const float* __restrict__ feat,
                          const _Float16* __restrict__ f1h, const float* __restrict__ b1,
                          const _Float16* __restrict__ f2h, const float* __restrict__ b2,
                          float* __restrict__ out) {
    __shared__ __align__(16) _Float16 A1[64 * 128];
    __shared__ __align__(16) _Float16 A2[64 * 256];
    int tid = threadIdx.x, lane = tid & 31, wv = tid >> 5;
    int m = lane & 15, hiHalf = lane >> 4;

    for (int i = tid; i < 64 * 128; i += 256) A1[i] = (_Float16)feat[i];
    __syncthreads();

    // GEMM1: (64x128) x (128x256), tiles: mt in [0,4), nt in [0,16)
    for (int tix = wv; tix < 64; tix += 8) {
        int mt = tix >> 4, nt = tix & 15;
        v8f acc = zero8f();
        const _Float16* arow = &A1[(mt * 16 + m) * 128];
        const _Float16* wrow = f1h + (size_t)(nt * 16 + m) * 128;
#pragma unroll
        for (int kc = 0; kc < 4; ++kc) {
            int ka = kc * 32 + (hiHalf ? 8 : 0);
            int kb = kc * 32 + (hiHalf ? 8 : 0);
            v8h alo = load8h(arow + ka);
            v8h ahi = load8h(arow + ka + 16);
            v8h blo = load8h(wrow + kb);
            v8h bhi = load8h(wrow + kb + 16);
            acc = wmma_f16(mk16(alo, ahi), mk16(blo, bhi), acc);
        }
        int n = nt * 16 + m;
        float bb = b1[n];
#pragma unroll
        for (int i = 0; i < 8; ++i)
            A2[(mt * 16 + hiHalf * 8 + i) * 256 + n] =
                (_Float16)fmaxf(acc[i] + bb, 0.0f);
    }
    __syncthreads();

    // GEMM2: (64x256) x (256x128), tiles: mt in [0,4), nt in [0,8)
    for (int tix = wv; tix < 32; tix += 8) {
        int mt = tix >> 3, nt = tix & 7;
        v8f acc = zero8f();
        const _Float16* arow = &A2[(mt * 16 + m) * 256];
        const _Float16* wrow = f2h + (size_t)(nt * 16 + m) * 256;
#pragma unroll
        for (int kc = 0; kc < 8; ++kc) {
            int ka = kc * 32 + (hiHalf ? 8 : 0);
            int kb = kc * 32 + (hiHalf ? 8 : 0);
            v8h alo = load8h(arow + ka);
            v8h ahi = load8h(arow + ka + 16);
            v8h blo = load8h(wrow + kb);
            v8h bhi = load8h(wrow + kb + 16);
            acc = wmma_f16(mk16(alo, ahi), mk16(blo, bhi), acc);
        }
        int n = nt * 16 + m;
        float bb = b2[n];
#pragma unroll
        for (int i = 0; i < 8; ++i)
            out[(size_t)(mt * 16 + hiHalf * 8 + i) * 128 + n] =
                fmaxf(acc[i] + bb, 0.0f);
    }
}

// ---------------------------------------------------------------------------
// Workspace layout (bytes). FFT buffer is reused for h1 after the front-end.
// ---------------------------------------------------------------------------
#define OFF_FFT   ((size_t)0)               // float2[64*512*512]  = 128 MB
#define OFF_H1    ((size_t)0)               // f16 [64*128*128*32] =  64 MB (reuse)
#define OFF_H2    ((size_t)67108864)        // f16 [64*32*32*64]   =   8 MB
#define OFF_FEAT  ((size_t)75497472)        // f32 [64*128]
#define OFF_LOGM  ((size_t)134217728)       // f32 [64*512*512]    =  64 MB
#define OFF_STATS ((size_t)201326592)       // f32 [128]
#define OFF_W2P   ((size_t)201327104)       // f16 [9*64*32]
#define OFF_W3P   ((size_t)201363968)       // f16 [9*128*64]
#define OFF_FC1H  ((size_t)201511424)       // f16 [256*128]
#define OFF_FC2H  ((size_t)201576960)       // f16 [128*256]

extern "C" void kernel_launch(void* const* d_in, const int* in_sizes, int n_in,
                              void* d_out, int out_size, void* d_ws, size_t ws_size,
                              hipStream_t stream) {
    (void)in_sizes; (void)n_in; (void)out_size; (void)ws_size;
    const float* x       = (const float*)d_in[0];
    const float* conv1_w = (const float*)d_in[1];
    const float* conv1_b = (const float*)d_in[2];
    const float* bn1_g   = (const float*)d_in[3];
    const float* bn1_b   = (const float*)d_in[4];
    const float* bn1_m   = (const float*)d_in[5];
    const float* bn1_v   = (const float*)d_in[6];
    const float* conv2_w = (const float*)d_in[7];
    const float* conv2_b = (const float*)d_in[8];
    const float* bn2_g   = (const float*)d_in[9];
    const float* bn2_b   = (const float*)d_in[10];
    const float* bn2_m   = (const float*)d_in[11];
    const float* bn2_v   = (const float*)d_in[12];
    const float* conv3_w = (const float*)d_in[13];
    const float* conv3_b = (const float*)d_in[14];
    const float* bn3_g   = (const float*)d_in[15];
    const float* bn3_b   = (const float*)d_in[16];
    const float* bn3_m   = (const float*)d_in[17];
    const float* bn3_v   = (const float*)d_in[18];
    const float* fc1_w   = (const float*)d_in[19];
    const float* fc1_b   = (const float*)d_in[20];
    const float* fc2_w   = (const float*)d_in[21];
    const float* fc2_b   = (const float*)d_in[22];

    char* ws = (char*)d_ws;
    float2*    F     = (float2*)(ws + OFF_FFT);
    _Float16*  h1    = (_Float16*)(ws + OFF_H1);
    _Float16*  h2    = (_Float16*)(ws + OFF_H2);
    float*     feat  = (float*)(ws + OFF_FEAT);
    float*     logm  = (float*)(ws + OFF_LOGM);
    float*     stats = (float*)(ws + OFF_STATS);
    _Float16*  w2p   = (_Float16*)(ws + OFF_W2P);
    _Float16*  w3p   = (_Float16*)(ws + OFF_W3P);
    _Float16*  fc1h  = (_Float16*)(ws + OFF_FC1H);
    _Float16*  fc2h  = (_Float16*)(ws + OFF_FC2H);
    float*     outp  = (float*)d_out;

    prep_kernel<<<288, 256, 0, stream>>>(conv2_w, conv3_w, fc1_w, fc2_w,
                                         w2p, w3p, fc1h, fc2h, stats, feat);
    fft_rows_kernel<<<64 * 512, 256, 0, stream>>>(x, F);
    fft_cols_kernel<<<64 * 512, 256, 0, stream>>>(F, logm, stats);
    conv1_kernel<<<64 * 64, 256, 0, stream>>>(logm, stats, conv1_w, conv1_b,
                                              bn1_g, bn1_b, bn1_m, bn1_v, h1);
    conv2_kernel<<<8192, 256, 0, stream>>>(h1, w2p, conv2_b,
                                           bn2_g, bn2_b, bn2_m, bn2_v, h2);
    conv3_kernel<<<1024, 256, 0, stream>>>(h2, w3p, conv3_b,
                                           bn3_g, bn3_b, bn3_m, bn3_v, feat);
    fc_kernel<<<1, 256, 0, stream>>>(feat, fc1h, fc1_b, fc2h, fc2_b, outp);
}
